// SelfAttention_3324304687128
// MI455X (gfx1250) — matrix operand order repaired and checked by reading, not compiled, not run on hardware
//
#include <hip/hip_runtime.h>
#include <hip/hip_bf16.h>

// ---------------------------------------------------------------------------
// SelfAttention block for MI455X (gfx1250, wave32, WMMA).
// GroupNorm -> QKV GEMM -> flash attention -> out-proj GEMM (+bias+residual).
// Matrix math: v_wmma_f32_16x16x32_f16 (f32 accum).
// Data movement: global_load_async_to_lds_b128 (ASYNCcnt DMA) with LDS
// double-buffering in the GEMMs, global_prefetch_b8 for L2 warm-up.
// ---------------------------------------------------------------------------

typedef __attribute__((ext_vector_type(16))) _Float16 v16h;
typedef __attribute__((ext_vector_type(8)))  _Float16 v8h;
typedef __attribute__((ext_vector_type(4)))  _Float16 v4h;
typedef __attribute__((ext_vector_type(8)))  float    v8f;

#define BATCH  4
#define SEQ    1024
#define CH     1024
#define NH     16
#define DH     64
#define GROUPS 32

#define LDT 40   // padded LDS stride (halves) for GEMM tiles
#define LQ  72   // padded LDS stride (halves) for attention tiles

__device__ __forceinline__ v8f wmma_f16(v16h a, v16h b, v8f c) {
  // (neg_a, A, neg_b, B, c_mod, C, reuse_a, reuse_b)
  return __builtin_amdgcn_wmma_f32_16x16x32_f16(false, a, false, b, (short)0, c,
                                                false, false);
}

// ---- gfx1250 async global->LDS DMA (ASYNCcnt) ------------------------------
// VDST = per-lane LDS byte address (low 32 bits of the flat shared address),
// VADDR = 64-bit global address, GV mode, 16 bytes per lane.
__device__ __forceinline__ void async_ld_b128(const void* g, void* l) {
  unsigned       lds  = (unsigned)(unsigned long long)(uintptr_t)l;
  unsigned long long ga = (unsigned long long)(uintptr_t)g;
  asm volatile("global_load_async_to_lds_b128 %0, %1, off"
               :: "v"(lds), "v"(ga)
               : "memory");
}
__device__ __forceinline__ void wait_async0() {
  asm volatile("s_wait_asynccnt 0x0" ::: "memory");
}

// Assemble a 16-half fragment from two contiguous 8-half LDS chunks.
__device__ __forceinline__ v16h frag_join(const _Float16* p0, const _Float16* p1) {
  v8h lo = *(const v8h*)p0;
  v8h hi = *(const v8h*)p1;
  v16h r;
#pragma unroll
  for (int i = 0; i < 8; ++i) { r[i] = lo[i]; r[i + 8] = hi[i]; }
  return r;
}

// A-matrix 16x32 fragment (ISA 7.12.2): lane L holds row (L&15);
// elems 0..7 = K[8*(L>>4)+j], elems 8..15 = K[16+8*(L>>4)+j].
__device__ __forceinline__ v16h frag_a(const _Float16* rowbase, int h16) {
  return frag_join(rowbase + 8 * h16, rowbase + 16 + 8 * h16);
}
// B-matrix 32x16 fragment: lane L holds col (L&15);
// elems 0..7 = K[8*(L>>4)+j], elems 8..15 = K[16+8*(L>>4)+j] (stored [n][k]).
__device__ __forceinline__ v16h frag_b(const _Float16* colbase, int h16) {
  return frag_join(colbase + 8 * h16, colbase + 16 + 8 * h16);
}

// ---------------------------------------------------------------------------
// f32 -> f16 conversion (weights)
// ---------------------------------------------------------------------------
__global__ __launch_bounds__(256) void cvt_f32_f16(const float* __restrict__ s,
                                                   _Float16* __restrict__ d,
                                                   int n4) {
  int i = blockIdx.x * 256 + threadIdx.x;
  if (i < n4) {
    float4 v = ((const float4*)s)[i];
    v4h o;
    o[0] = (_Float16)v.x; o[1] = (_Float16)v.y;
    o[2] = (_Float16)v.z; o[3] = (_Float16)v.w;
    ((v4h*)d)[i] = o;
  }
}

// ---------------------------------------------------------------------------
// GroupNorm over dim1 groups: block = one (batch, group) of 32*1024 floats.
// ---------------------------------------------------------------------------
__global__ __launch_bounds__(256) void groupnorm_kernel(
    const float* __restrict__ x, const float* __restrict__ w,
    const float* __restrict__ b, _Float16* __restrict__ h) {
  const int tid = threadIdx.x;
  const int blk = blockIdx.x;        // b*GROUPS + g
  const int bb  = blk >> 5;
  const int g   = blk & 31;
  const int NELEM = (SEQ / GROUPS) * CH;  // 32768
  const float* xp = x + (size_t)bb * SEQ * CH + (size_t)g * (SEQ / GROUPS) * CH;

  float sum = 0.f, sq = 0.f;
  for (int i = tid; i < NELEM / 4; i += 256) {
    float4 v = ((const float4*)xp)[i];
    sum += v.x + v.y + v.z + v.w;
    sq  += v.x * v.x + v.y * v.y + v.z * v.z + v.w * v.w;
  }
  __shared__ float s1[256], s2[256];
  s1[tid] = sum; s2[tid] = sq;
  __syncthreads();
  for (int st = 128; st > 0; st >>= 1) {
    if (tid < st) { s1[tid] += s1[tid + st]; s2[tid] += s2[tid + st]; }
    __syncthreads();
  }
  const float mu   = s1[0] * (1.0f / NELEM);
  const float var  = s2[0] * (1.0f / NELEM) - mu * mu;
  const float rstd = rsqrtf(var + 1e-6f);

  _Float16* hp = h + (size_t)bb * SEQ * CH + (size_t)g * (SEQ / GROUPS) * CH;
  for (int i = tid; i < NELEM; i += 256) {
    int srow = g * (SEQ / GROUPS) + (i >> 10);
    float xn = (xp[i] - mu) * rstd;
    hp[i] = (_Float16)(xn * w[srow] + b[srow]);
  }
}

// ---------------------------------------------------------------------------
// QKV GEMM: (4096x1024 f16) x (w_qkv[n][k] f16) + b_qkv.
// 128x128 block tile, 8 waves (4x2), each wave 2x4 16x16 WMMA tiles.
// Async-DMA double-buffered LDS; one barrier per K step.
// Epilogue scatters Q,K -> [b,h,s,d] and V -> [b,h,d,s] (transposed).
// ---------------------------------------------------------------------------
__global__ __launch_bounds__(256) void gemm_qkv_kernel(
    const _Float16* __restrict__ Ag, const _Float16* __restrict__ Bg,
    const float* __restrict__ bias, _Float16* __restrict__ qO,
    _Float16* __restrict__ kO, _Float16* __restrict__ vtO) {
  __shared__ _Float16 lA[2][128 * LDT];
  __shared__ _Float16 lB[2][128 * LDT];

  const int tid  = threadIdx.x;
  const int lane = tid & 31;
  const int wave = tid >> 5;
  const int wm = wave >> 1, wn = wave & 1;
  const int l16 = lane & 15, h16 = lane >> 4;
  const int m0 = blockIdx.y * 128;
  const int n0 = blockIdx.x * 128;
  const int row  = tid >> 2;          // 0..63
  const int offq = (tid & 3) * 8;     // 0,8,16,24 halves

  auto issue_tile = [&](int buf, int k0) {
    async_ld_b128(&Ag[(size_t)(m0 + row) * CH + k0 + offq],      &lA[buf][row * LDT + offq]);
    async_ld_b128(&Ag[(size_t)(m0 + row + 64) * CH + k0 + offq], &lA[buf][(row + 64) * LDT + offq]);
    async_ld_b128(&Bg[(size_t)(n0 + row) * CH + k0 + offq],      &lB[buf][row * LDT + offq]);
    async_ld_b128(&Bg[(size_t)(n0 + row + 64) * CH + k0 + offq], &lB[buf][(row + 64) * LDT + offq]);
  };

  v8f zf = {};
  v8f acc[2][4];
#pragma unroll
  for (int mt = 0; mt < 2; ++mt)
#pragma unroll
    for (int nt = 0; nt < 4; ++nt) acc[mt][nt] = zf;

  issue_tile(0, 0);
  for (int k0 = 0; k0 < CH; k0 += 32) {
    const int cur = (k0 >> 5) & 1;
    wait_async0();                 // my current-tile DMAs are in LDS
    __syncthreads();               // everyone's are; prev reads of buf cur^1 done
    if (k0 + 32 < CH) issue_tile(cur ^ 1, k0 + 32);
    if (k0 + 64 < CH) {            // warm L2 two tiles ahead
      __builtin_prefetch(&Ag[(size_t)(m0 + row) * CH + k0 + 64 + offq], 0, 1);
      __builtin_prefetch(&Bg[(size_t)(n0 + row) * CH + k0 + 64 + offq], 0, 1);
    }

    v16h af[2], bf[4];
#pragma unroll
    for (int mt = 0; mt < 2; ++mt)
      af[mt] = frag_a(&lA[cur][(wm * 32 + mt * 16 + l16) * LDT], h16);
#pragma unroll
    for (int nt = 0; nt < 4; ++nt)
      bf[nt] = frag_b(&lB[cur][(wn * 64 + nt * 16 + l16) * LDT], h16);
#pragma unroll
    for (int mt = 0; mt < 2; ++mt)
#pragma unroll
      for (int nt = 0; nt < 4; ++nt)
        acc[mt][nt] = wmma_f16(af[mt], bf[nt], acc[mt][nt]);
  }

#pragma unroll
  for (int mt = 0; mt < 2; ++mt)
#pragma unroll
    for (int nt = 0; nt < 4; ++nt)
#pragma unroll
      for (int vv = 0; vv < 8; ++vv) {
        int m = m0 + wm * 32 + mt * 16 + vv + 8 * h16;
        int n = n0 + wn * 64 + nt * 16 + l16;
        float val = acc[mt][nt][vv] + bias[n];
        int part = n >> 10, cc = n & 1023;
        int head = cc >> 6, d = cc & 63;
        int bb = m >> 10, s = m & 1023;
        _Float16 hv = (_Float16)val;
        size_t bh = (size_t)(bb * NH + head);
        if (part == 0)      qO[(bh * SEQ + s) * DH + d] = hv;
        else if (part == 1) kO[(bh * SEQ + s) * DH + d] = hv;
        else                vtO[(bh * DH + d) * SEQ + s] = hv;
      }
}

// ---------------------------------------------------------------------------
// Flash attention: block = 64 query rows of one (b,h); 4 waves, each wave owns
// 16 rows x full 64-col score/output tiles. Streams 16 key blocks of 64.
// Tile loads via async global->LDS DMA. Output written in the reference's
// raw-reshape order == [b,h,s,d] flat.
// ---------------------------------------------------------------------------
__global__ __launch_bounds__(128) void attn_kernel(
    const _Float16* __restrict__ q, const _Float16* __restrict__ k,
    const _Float16* __restrict__ vT, _Float16* __restrict__ o) {
  __shared__ _Float16 lQ[64 * LQ];
  __shared__ _Float16 lK[64 * LQ];
  __shared__ _Float16 lV[64 * LQ];   // V transposed: [d][key]
  __shared__ _Float16 lP[64 * LQ];   // softmax probs, per-wave 16-row slices

  const int tid  = threadIdx.x;
  const int lane = tid & 31;
  const int wave = tid >> 5;
  const int l16 = lane & 15, h16 = lane >> 4;
  const int blk = blockIdx.x;
  const int qb  = blk & (SEQ / 64 - 1);    // 0..15
  const int bh  = blk >> 4;                // 0..63
  const int q0  = qb * 64;

  const _Float16* Qg = q  + (size_t)bh * SEQ * DH + (size_t)q0 * DH;
  const _Float16* Kg = k  + (size_t)bh * SEQ * DH;
  const _Float16* Vg = vT + (size_t)bh * DH * SEQ;

#pragma unroll
  for (int it = 0; it < 4; ++it) {
    int t = tid + it * 128;                // 0..511, 8 halves each
    int r = t >> 2, of = (t & 3) * 8;
    async_ld_b128(&Qg[r * DH + of], &lQ[r * LQ + of]);
  }

  v8f zf = {};
  v8f oacc[4];
  float mrow[8], lrow[8];
#pragma unroll
  for (int t = 0; t < 4; ++t) oacc[t] = zf;
#pragma unroll
  for (int vv = 0; vv < 8; ++vv) { mrow[vv] = -1e30f; lrow[vv] = 0.f; }

  const float sc = 1.0f / (float)DH;       // faithful double-sqrt scale

  for (int j = 0; j < SEQ / 64; ++j) {
    __syncthreads();                       // prior reads of lK/lV done
#pragma unroll
    for (int it = 0; it < 4; ++it) {
      int t = tid + it * 128;
      int r = t >> 2, of = (t & 3) * 8;
      async_ld_b128(&Kg[(size_t)(j * 64 + r) * DH + of], &lK[r * LQ + of]);
      async_ld_b128(&Vg[(size_t)r * SEQ + j * 64 + of],  &lV[r * LQ + of]);
    }
    wait_async0();                         // my K/V (and Q on j==0) DMAs done
    __syncthreads();                       // everyone's done

    // S = Q(16x64) . K^T(64x64) for this wave's row slice
    v8f st[4];
#pragma unroll
    for (int nt = 0; nt < 4; ++nt) st[nt] = zf;
#pragma unroll
    for (int kc = 0; kc < 2; ++kc) {
      v16h aq = frag_a(&lQ[(wave * 16 + l16) * LQ + kc * 32], h16);
#pragma unroll
      for (int nt = 0; nt < 4; ++nt) {
        v16h bk = frag_b(&lK[(nt * 16 + l16) * LQ + kc * 32], h16);
        st[nt] = wmma_f16(aq, bk, st[nt]);
      }
    }

    // online softmax across the 64 key columns of this block
#pragma unroll
    for (int vv = 0; vv < 8; ++vv) {
      float rm = -1e30f;
#pragma unroll
      for (int nt = 0; nt < 4; ++nt) rm = fmaxf(rm, st[nt][vv]);
      rm *= sc;
#pragma unroll
      for (int off = 8; off >= 1; off >>= 1) rm = fmaxf(rm, __shfl_xor(rm, off, 16));
      float mn = fmaxf(mrow[vv], rm);
      float alpha = __expf(mrow[vv] - mn);
      mrow[vv] = mn;
      float rs = 0.f;
#pragma unroll
      for (int nt = 0; nt < 4; ++nt) {
        float p = __expf(st[nt][vv] * sc - mn);
        st[nt][vv] = p;
        rs += p;
      }
#pragma unroll
      for (int off = 8; off >= 1; off >>= 1) rs += __shfl_xor(rs, off, 16);
      lrow[vv] = lrow[vv] * alpha + rs;
#pragma unroll
      for (int t = 0; t < 4; ++t) oacc[t][vv] *= alpha;
    }

    // re-layout P (C/D frag) -> A frag via LDS
#pragma unroll
    for (int nt = 0; nt < 4; ++nt)
#pragma unroll
      for (int vv = 0; vv < 8; ++vv)
        lP[(wave * 16 + vv + 8 * h16) * LQ + nt * 16 + l16] = (_Float16)st[nt][vv];
    __syncthreads();

    // O += P(16x64) . V(64x64)
#pragma unroll
    for (int kc = 0; kc < 2; ++kc) {
      v16h ap = frag_a(&lP[(wave * 16 + l16) * LQ + kc * 32], h16);
#pragma unroll
      for (int t = 0; t < 4; ++t) {
        v16h bv = frag_b(&lV[(t * 16 + l16) * LQ + kc * 32], h16);
        oacc[t] = wmma_f16(ap, bv, oacc[t]);
      }
    }
  }

  // normalize and store O in [b,h,s,d] flat order (= raw reshape order)
#pragma unroll
  for (int t = 0; t < 4; ++t)
#pragma unroll
    for (int vv = 0; vv < 8; ++vv) {
      int r = wave * 16 + vv + 8 * h16;
      float val = oacc[t][vv] / lrow[vv];
      o[(size_t)bh * SEQ * DH + (size_t)(q0 + r) * DH + t * 16 + l16] = (_Float16)val;
    }
}

// ---------------------------------------------------------------------------
// Out projection: (4096x1024 f16) x (w_out[n][k] f16) + b_out + residual x.
// Same async double-buffered WMMA GEMM.
// ---------------------------------------------------------------------------
__global__ __launch_bounds__(256) void gemm_out_kernel(
    const _Float16* __restrict__ Ag, const _Float16* __restrict__ Bg,
    const float* __restrict__ bias, const float* __restrict__ xres,
    float* __restrict__ out) {
  __shared__ _Float16 lA[2][128 * LDT];
  __shared__ _Float16 lB[2][128 * LDT];

  const int tid  = threadIdx.x;
  const int lane = tid & 31;
  const int wave = tid >> 5;
  const int wm = wave >> 1, wn = wave & 1;
  const int l16 = lane & 15, h16 = lane >> 4;
  const int m0 = blockIdx.y * 128;
  const int n0 = blockIdx.x * 128;
  const int row  = tid >> 2;
  const int offq = (tid & 3) * 8;

  auto issue_tile = [&](int buf, int k0) {
    async_ld_b128(&Ag[(size_t)(m0 + row) * CH + k0 + offq],      &lA[buf][row * LDT + offq]);
    async_ld_b128(&Ag[(size_t)(m0 + row + 64) * CH + k0 + offq], &lA[buf][(row + 64) * LDT + offq]);
    async_ld_b128(&Bg[(size_t)(n0 + row) * CH + k0 + offq],      &lB[buf][row * LDT + offq]);
    async_ld_b128(&Bg[(size_t)(n0 + row + 64) * CH + k0 + offq], &lB[buf][(row + 64) * LDT + offq]);
  };

  v8f zf = {};
  v8f acc[2][4];
#pragma unroll
  for (int mt = 0; mt < 2; ++mt)
#pragma unroll
    for (int nt = 0; nt < 4; ++nt) acc[mt][nt] = zf;

  issue_tile(0, 0);
  for (int k0 = 0; k0 < CH; k0 += 32) {
    const int cur = (k0 >> 5) & 1;
    wait_async0();
    __syncthreads();
    if (k0 + 32 < CH) issue_tile(cur ^ 1, k0 + 32);
    if (k0 + 64 < CH) {
      __builtin_prefetch(&Ag[(size_t)(m0 + row) * CH + k0 + 64 + offq], 0, 1);
      __builtin_prefetch(&Bg[(size_t)(n0 + row) * CH + k0 + 64 + offq], 0, 1);
    }

    v16h af[2], bf[4];
#pragma unroll
    for (int mt = 0; mt < 2; ++mt)
      af[mt] = frag_a(&lA[cur][(wm * 32 + mt * 16 + l16) * LDT], h16);
#pragma unroll
    for (int nt = 0; nt < 4; ++nt)
      bf[nt] = frag_b(&lB[cur][(wn * 64 + nt * 16 + l16) * LDT], h16);
#pragma unroll
    for (int mt = 0; mt < 2; ++mt)
#pragma unroll
      for (int nt = 0; nt < 4; ++nt)
        acc[mt][nt] = wmma_f16(af[mt], bf[nt], acc[mt][nt]);
  }

#pragma unroll
  for (int mt = 0; mt < 2; ++mt)
#pragma unroll
    for (int nt = 0; nt < 4; ++nt)
#pragma unroll
      for (int vv = 0; vv < 8; ++vv) {
        int m = m0 + wm * 32 + mt * 16 + vv + 8 * h16;
        int n = n0 + wn * 64 + nt * 16 + l16;
        size_t idx = (size_t)m * CH + n;
        out[idx] = acc[mt][nt][vv] + bias[n] + xres[idx];
      }
}

// ---------------------------------------------------------------------------
// Launch
// ---------------------------------------------------------------------------
extern "C" void kernel_launch(void* const* d_in, const int* in_sizes, int n_in,
                              void* d_out, int out_size, void* d_ws, size_t ws_size,
                              hipStream_t stream) {
  const float* x    = (const float*)d_in[0];
  const float* gnw  = (const float*)d_in[1];
  const float* gnb  = (const float*)d_in[2];
  const float* wqkv = (const float*)d_in[3];
  const float* bqkv = (const float*)d_in[4];
  const float* wout = (const float*)d_in[5];
  const float* bout = (const float*)d_in[6];
  float* out = (float*)d_out;

  char* ws = (char*)d_ws;
  size_t off = 0;
  auto alloc = [&](size_t bytes) -> void* {
    void* p = ws + off;
    off = (off + bytes + 255) & ~(size_t)255;
    return p;
  };

  const size_t actH = (size_t)BATCH * SEQ * CH * sizeof(_Float16);  // 8 MB
  _Float16* hH    = (_Float16*)alloc(actH);                    // GN out; reused as attn out
  _Float16* wqkvH = (_Float16*)alloc((size_t)3 * CH * CH * sizeof(_Float16));
  _Float16* woutH = (_Float16*)alloc((size_t)CH * CH * sizeof(_Float16));
  _Float16* qH    = (_Float16*)alloc(actH);
  _Float16* kH    = (_Float16*)alloc(actH);
  _Float16* vtH   = (_Float16*)alloc(actH);

  {
    int n4 = 3 * CH * CH / 4;
    cvt_f32_f16<<<(n4 + 255) / 256, 256, 0, stream>>>(wqkv, wqkvH, n4);
  }
  {
    int n4 = CH * CH / 4;
    cvt_f32_f16<<<(n4 + 255) / 256, 256, 0, stream>>>(wout, woutH, n4);
  }
  groupnorm_kernel<<<BATCH * GROUPS, 256, 0, stream>>>(x, gnw, gnb, hH);
  gemm_qkv_kernel<<<dim3(3 * CH / 128, BATCH * SEQ / 128), 256, 0, stream>>>(
      hH, wqkvH, bqkv, qH, kH, vtH);
  attn_kernel<<<BATCH * NH * (SEQ / 64), 128, 0, stream>>>(qH, kH, vtH, hH);
  gemm_out_kernel<<<dim3(CH / 128, BATCH * SEQ / 128), 256, 0, stream>>>(
      hH, woutH, bout, x, out);
}
